// EnsembleDeepSDF_64905545777862
// MI455X (gfx1250) — compile-verified, matrix operand order checked by reading, not run on hardware
//
#include <hip/hip_runtime.h>
#include <hip/hip_bf16.h>

typedef __attribute__((ext_vector_type(16))) _Float16 v16h;
typedef __attribute__((ext_vector_type(8)))  _Float16 v8h;
typedef __attribute__((ext_vector_type(8)))  float    v8f;
typedef __attribute__((ext_vector_type(4)))  int      v4i;
typedef __attribute__((ext_vector_type(8)))  int      v8i;

#define T_NUM 9
#define D_IN 67
#define K0P 96          // padded K for layer 0
#define H 512
#define NHID 7
#define NPTS 32768
#define ROWS 64         // rows per workgroup
#define THREADS 256     // 8 wave32

#define W0H_ELEMS (T_NUM * H * K0P)            // 442368 halves
#define WHH_ELEMS (T_NUM * NHID * H * H)       // 16515072 halves

#define AS1 __attribute__((address_space(1)))
#define AS3 __attribute__((address_space(3)))

// Toggle: 1 = use inline-asm Tensor Data Mover for weight staging,
//         0 = fall back to async-to-LDS / register staging.
#define USE_TDM_ASM 1

#if defined(__has_builtin)
#if __has_builtin(__builtin_amdgcn_global_load_async_to_lds_b128) && \
    __has_builtin(__builtin_amdgcn_s_wait_asynccnt)
#define HAVE_ASYNC_LDS 1
#endif
#endif

#ifdef HAVE_ASYNC_LDS
#define ASYNC_CP16(gp, lp) \
    __builtin_amdgcn_global_load_async_to_lds_b128((AS1 v4i*)(gp), (AS3 v4i*)(lp), 0, 0)
#define WAIT_ASYNC() __builtin_amdgcn_s_wait_asynccnt(0)
#endif

#if USE_TDM_ASM
#define HAVE_TDM 1
#if defined(__has_builtin) && __has_builtin(__builtin_amdgcn_s_wait_tensorcnt)
#define WAIT_TENSOR() __builtin_amdgcn_s_wait_tensorcnt(0)
#else
#define WAIT_TENSOR() asm volatile("s_wait_tensorcnt 0x0" ::: "memory")
#endif

// byte offset of a generic pointer within the workgroup's LDS allocation
__device__ __forceinline__ unsigned lds_off(const void* p) {
    return (unsigned)(unsigned long long)(AS3 const char*)(const char*)p;
}

// Issue one TDM 2D tile load: tileH rows of tileW f16 elements, global row
// stride rowStride elements, stored contiguously at LDS byte offset ldsDst.
// D# group0/group1 packing per CDNA5 ISA "Tensor DMA Descriptor" 8.3/8.4;
// 2-group form (VADDR2/3 NULL) handles tensors up to 2D.
__device__ __forceinline__ void tdm_load_2d(const void* gsrc, unsigned ldsDst,
                                            unsigned tileW, unsigned tileH,
                                            unsigned rowStride) {
    unsigned long long ga = (unsigned long long)gsrc;
    v4i g0;
    g0[0] = 1;                                         // count=1 (valid user D#)
    g0[1] = (int)ldsDst;                               // lds_addr (bytes)
    g0[2] = (int)(unsigned)(ga & 0xFFFFFFFFu);         // global_addr[31:0]
    g0[3] = (int)(((unsigned)(ga >> 32) & 0x01FFFFFFu) // global_addr[56:32]
                  | 0x80000000u);                      // type=2 ("image")
    v8i g1;
    g1[0] = (int)(1u << 16);                           // data_size=1 -> 2 bytes
    g1[1] = (int)(tileW << 16);                        // tensor_dim0[15:0]
    g1[2] = (int)((tileW >> 16) | (tileH << 16));      // tdim0[31:16]|tdim1[15:0]
    g1[3] = (int)((tileH >> 16) | (tileW << 16));      // tdim1[31:16]|tile_dim0
    g1[4] = (int)(tileH & 0xFFFFu);                    // tile_dim1 (tile_dim2=0)
    g1[5] = (int)rowStride;                            // tensor_dim0_stride[31:0]
    g1[6] = 0;
    g1[7] = 0;
    asm volatile("tensor_load_to_lds %0, %1" :: "s"(g0), "s"(g1) : "memory");
}
#endif // USE_TDM_ASM

// ---------------------------------------------------------------------------
// Weight conversion: f32 -> f16, W0 zero-padded from K=67 to K=96.
// ---------------------------------------------------------------------------
__global__ void convert_weights_kernel(const float* __restrict__ W0,
                                       const float* __restrict__ Wh,
                                       _Float16* __restrict__ W0h,
                                       _Float16* __restrict__ Whh) {
    const int total0 = W0H_ELEMS;
    const int total1 = WHH_ELEMS;
    for (int idx = blockIdx.x * blockDim.x + threadIdx.x;
         idx < total0 + total1; idx += gridDim.x * blockDim.x) {
        if (idx < total0) {
            int t = idx / (H * K0P);
            int r = idx - t * (H * K0P);
            int n = r / K0P;
            int k = r - n * K0P;
            float v = (k < D_IN) ? W0[(t * H + n) * D_IN + k] : 0.f;
            W0h[idx] = (_Float16)v;
        } else {
            int j = idx - total0;
            Whh[j] = (_Float16)Wh[j];
        }
    }
}

// ---------------------------------------------------------------------------
// Helpers
// ---------------------------------------------------------------------------
__device__ __forceinline__ float softplus_beta100(float x) {
    float z = 100.f * x;
    float e = __expf(z);
    float r = __logf(1.f + e) * 0.01f;
    return (z > 20.f) ? x : r;   // large z: softplus(z)/100 == x
}

// Per ISA 7.12.2, each lane's half of an f16 fragment is two contiguous
// 8-half (16B) chunks at K offsets {0,+16} (upper lanes shifted by +8).
__device__ __forceinline__ v16h ldfrag(const _Float16* p) {
    v8h lo = *(const v8h*)(p);
    v8h hi = *(const v8h*)(p + 16);
    return __builtin_shufflevector(lo, hi, 0, 1, 2, 3, 4, 5, 6, 7,
                                           8, 9, 10, 11, 12, 13, 14, 15);
}

// ---------------------------------------------------------------------------
// Main MoE-MLP kernel: one workgroup = 64 rows, 8 waves.
// Wave w owns columns [w*64, w*64+64) (4 N-tiles) x all 4 M-tiles.
// ---------------------------------------------------------------------------
__global__ __launch_bounds__(THREADS)
void ensemble_mlp_kernel(const float* __restrict__ x,
                         const int* __restrict__ tv,
                         const float* __restrict__ b0,
                         const float* __restrict__ bh,
                         const float* __restrict__ Wo,
                         const float* __restrict__ bo,
                         const _Float16* __restrict__ W0h,
                         const _Float16* __restrict__ Whh,
                         float* __restrict__ out) {
    // LDS: h ping (64x512 f16) + h pong + weight slab double buffer (2 x 512x64)
    __shared__ __attribute__((aligned(16))) _Float16 smem[2 * ROWS * H + 2 * H * 64];
    _Float16* sh_h0 = smem;                    // 32768 halves
    _Float16* sh_h1 = smem + ROWS * H;         // 32768 halves
    _Float16* sh_w  = smem + 2 * ROWS * H;     // 65536 halves (layer0: 512x96 fits)

    const int tid   = threadIdx.x;
    const int lane  = tid & 31;
    const int wave  = tid >> 5;
    const int nBase = wave * 64;               // 4 N-tiles: nBase + i*16
    const int laneM = lane & 15;
    const int laneK = (lane & 16) ? 8 : 0;     // K / D-row offset for hi lanes
    const int rowg0 = blockIdx.x * ROWS;

    const int t0 = tv[rowg0];
    const int t1 = tv[rowg0 + ROWS - 1];

    for (int t = t0; t <= t1; ++t) {
        // ---- stage W0h[t] (512x96 halves = 96KB, contiguous) -> sh_w ----
        {
            const _Float16* w0g = W0h + t * (H * K0P);
#if defined(HAVE_TDM)
            if (wave == 0)
                tdm_load_2d(w0g, lds_off(sh_w), H * K0P, 1, H * K0P);
#elif defined(HAVE_ASYNC_LDS)
            for (int c = tid; c < (H * K0P) / 8; c += THREADS)
                ASYNC_CP16(w0g + c * 8, sh_w + c * 8);
#else
            for (int c = tid; c < (H * K0P) / 8; c += THREADS)
                *(uint4*)(sh_w + c * 8) = *(const uint4*)(w0g + c * 8);
#endif
        }
        // ---- stage x -> sh_h0 as f16 [64][96], zero padded ----
        for (int c = tid; c < ROWS * K0P; c += THREADS) {
            int r = c / K0P, k = c - r * K0P;
            float v = (k < D_IN) ? x[(rowg0 + r) * D_IN + k] : 0.f;
            sh_h0[c] = (_Float16)v;
        }
#if defined(HAVE_TDM)
        if (wave == 0) WAIT_TENSOR();
#elif defined(HAVE_ASYNC_LDS)
        WAIT_ASYNC();
#endif
        __syncthreads();

        v8f acc[4][4];

        // ================= layer 0: [64,96] x [96,512] =================
        #pragma unroll
        for (int m = 0; m < 4; ++m)
            #pragma unroll
            for (int i = 0; i < 4; ++i)
                #pragma unroll
                for (int j = 0; j < 8; ++j) acc[m][i][j] = 0.f;

        #pragma unroll
        for (int ks = 0; ks < 3; ++ks) {
            const int k0 = ks * 32;
            v16h a[4];
            #pragma unroll
            for (int m = 0; m < 4; ++m)
                a[m] = ldfrag(sh_h0 + (m * 16 + laneM) * K0P + k0 + laneK);
            #pragma unroll
            for (int i = 0; i < 4; ++i) {
                v16h b = ldfrag(sh_w + (nBase + i * 16 + laneM) * K0P + k0 + laneK);
                #pragma unroll
                for (int m = 0; m < 4; ++m)
                    acc[m][i] = __builtin_amdgcn_wmma_f32_16x16x32_f16(
                        false, a[m], false, b, (short)0, acc[m][i], false, false);
            }
        }
        // activation + bias -> sh_h1 [64][512]
        {
            const float* b0t = b0 + t * H;
            #pragma unroll
            for (int i = 0; i < 4; ++i) {
                int n = nBase + i * 16 + laneM;
                float bn = b0t[n];
                #pragma unroll
                for (int m = 0; m < 4; ++m)
                    #pragma unroll
                    for (int j = 0; j < 8; ++j) {
                        int row = m * 16 + j + laneK;
                        sh_h1[row * H + n] =
                            (_Float16)softplus_beta100(acc[m][i][j] + bn);
                    }
            }
        }
        __syncthreads();

        // ================= 7 hidden layers: [64,512] x [512,512] =========
        const _Float16* hin = sh_h1;
        _Float16* hout = sh_h0;
        for (int l = 0; l < NHID; ++l) {
            const _Float16* wg = Whh + (size_t)(t * NHID + l) * (H * H);

            #pragma unroll
            for (int m = 0; m < 4; ++m)
                #pragma unroll
                for (int i = 0; i < 4; ++i)
                    #pragma unroll
                    for (int j = 0; j < 8; ++j) acc[m][i][j] = 0.f;

            // preload K-slab 0 (512 rows x 64 halves = 64KB) into buffer 0
#if defined(HAVE_TDM)
            if (wave == 0) {
                tdm_load_2d(wg, lds_off(sh_w), 64, H, H);
                WAIT_TENSOR();
            }
#elif defined(HAVE_ASYNC_LDS)
            #pragma unroll
            for (int u = 0; u < 16; ++u) {
                int c = tid + u * THREADS, n = c >> 3, j = c & 7;
                ASYNC_CP16(wg + n * H + j * 8, sh_w + n * 64 + j * 8);
            }
            WAIT_ASYNC();
#else
            uint4 stage[16];
            #pragma unroll
            for (int u = 0; u < 16; ++u) {
                int c = tid + u * THREADS, n = c >> 3, j = c & 7;
                stage[u] = *(const uint4*)(wg + n * H + j * 8);
            }
            #pragma unroll
            for (int u = 0; u < 16; ++u) {
                int c = tid + u * THREADS, n = c >> 3, j = c & 7;
                *(uint4*)(sh_w + n * 64 + j * 8) = stage[u];
            }
#endif
            __syncthreads();

            for (int ks = 0; ks < 8; ++ks) {
                _Float16* cur = sh_w + (ks & 1) * (H * 64);
                _Float16* nxt = sh_w + ((ks + 1) & 1) * (H * 64);
                // software pipeline: start next slab's L2->LDS transfer, then
                // run WMMA on the current slab while it flies.
                if (ks < 7) {
#if defined(HAVE_TDM)
                    if (wave == 0)
                        tdm_load_2d(wg + (ks + 1) * 64, lds_off(nxt), 64, H, H);
#elif defined(HAVE_ASYNC_LDS)
                    #pragma unroll
                    for (int u = 0; u < 16; ++u) {
                        int c = tid + u * THREADS, n = c >> 3, j = c & 7;
                        ASYNC_CP16(wg + n * H + (ks + 1) * 64 + j * 8,
                                   nxt + n * 64 + j * 8);
                    }
#else
                    #pragma unroll
                    for (int u = 0; u < 16; ++u) {
                        int c = tid + u * THREADS, n = c >> 3, j = c & 7;
                        stage[u] = *(const uint4*)(wg + n * H + (ks + 1) * 64 + j * 8);
                    }
#endif
                }
                #pragma unroll
                for (int kk = 0; kk < 2; ++kk) {
                    const int k0g = ks * 64 + kk * 32;
                    v16h a[4];
                    #pragma unroll
                    for (int m = 0; m < 4; ++m)
                        a[m] = ldfrag(hin + (m * 16 + laneM) * H + k0g + laneK);
                    #pragma unroll
                    for (int i = 0; i < 4; ++i) {
                        v16h b = ldfrag(cur + (nBase + i * 16 + laneM) * 64 +
                                        kk * 32 + laneK);
                        #pragma unroll
                        for (int m = 0; m < 4; ++m)
                            acc[m][i] = __builtin_amdgcn_wmma_f32_16x16x32_f16(
                                false, a[m], false, b, (short)0, acc[m][i],
                                false, false);
                    }
                }
                if (ks < 7) {
#if defined(HAVE_TDM)
                    if (wave == 0) WAIT_TENSOR();
#elif defined(HAVE_ASYNC_LDS)
                    WAIT_ASYNC();
#else
                    #pragma unroll
                    for (int u = 0; u < 16; ++u) {
                        int c = tid + u * THREADS, n = c >> 3, j = c & 7;
                        *(uint4*)(nxt + n * 64 + j * 8) = stage[u];
                    }
#endif
                }
                __syncthreads();
            }

            // activation + bias -> hout
            {
                const float* bht = bh + (t * NHID + l) * H;
                #pragma unroll
                for (int i = 0; i < 4; ++i) {
                    int n = nBase + i * 16 + laneM;
                    float bn = bht[n];
                    #pragma unroll
                    for (int m = 0; m < 4; ++m)
                        #pragma unroll
                        for (int j = 0; j < 8; ++j) {
                            int row = m * 16 + j + laneK;
                            hout[row * H + n] =
                                (_Float16)softplus_beta100(acc[m][i][j] + bn);
                        }
                }
            }
            __syncthreads();
            _Float16* tmp = (_Float16*)hin; hin = hout; hout = tmp;
        }
        // after 7 swaps final activations are in sh_h0 (== hin)

        // ================= output layer: [64,512] x [512,1] ==============
        {
            const float* wot = Wo + t * H;
            float* pb = (float*)sh_w;           // 64x4 partials
            int row = tid >> 2, seg = tid & 3;
            const _Float16* hr = hin + row * H + seg * 128;
            const float* ws = wot + seg * 128;
            float s = 0.f;
            #pragma unroll 16
            for (int k = 0; k < 128; ++k) s += (float)hr[k] * ws[k];
            pb[(row << 2) + seg] = s;
            __syncthreads();
            if (tid < ROWS) {
                int rg = rowg0 + tid;
                if (tv[rg] == t) {
                    out[rg] = pb[tid * 4] + pb[tid * 4 + 1] +
                              pb[tid * 4 + 2] + pb[tid * 4 + 3] + bo[t];
                }
            }
            __syncthreads();   // smem reused next t iteration
        }
    }
}

extern "C" void kernel_launch(void* const* d_in, const int* in_sizes, int n_in,
                              void* d_out, int out_size, void* d_ws, size_t ws_size,
                              hipStream_t stream) {
    const float* x  = (const float*)d_in[0];
    const int*   tv = (const int*)d_in[1];
    const float* W0 = (const float*)d_in[2];
    const float* b0 = (const float*)d_in[3];
    const float* Wh = (const float*)d_in[4];
    const float* bh = (const float*)d_in[5];
    const float* Wo = (const float*)d_in[6];
    const float* bo = (const float*)d_in[7];
    float* out = (float*)d_out;

    _Float16* W0h = (_Float16*)d_ws;
    _Float16* Whh = W0h + W0H_ELEMS;

    convert_weights_kernel<<<2048, 256, 0, stream>>>(W0, Wh, W0h, Whh);
    ensemble_mlp_kernel<<<NPTS / ROWS, THREADS, 0, stream>>>(
        x, tv, b0, bh, Wo, bo, W0h, Whh, out);
}
